// SequenceReductionAttention_52329881535113
// MI455X (gfx1250) — matrix operand order repaired and checked by reading, not compiled, not run on hardware
//
#include <hip/hip_runtime.h>
#include <hip/hip_bf16.h>

// ---------------------------------------------------------------------------
// SequenceReductionAttention on MI455X (gfx1250, wave32, WMMA bf16)
// B=4, N=4096, C=256, HEADS=8, d=32, SR=2, H=W=64, N'=1024
// ---------------------------------------------------------------------------

typedef __attribute__((ext_vector_type(16))) __bf16 v16bf;
typedef __attribute__((ext_vector_type(8)))  __bf16 v8bf;
typedef __attribute__((ext_vector_type(8)))  float   v8f;

__device__ __forceinline__ __bf16 f2bf(float f) {
    unsigned u = __builtin_bit_cast(unsigned, f);
    u = (u + 0x7FFFu + ((u >> 16) & 1u)) >> 16;      // round-to-nearest-even
    unsigned short s = (unsigned short)u;
    return __builtin_bit_cast(__bf16, s);
}

__device__ __forceinline__ v8f wmma_bf16(v16bf a, v16bf b, v8f c) {
    // v_wmma_f32_16x16x32_bf16
    return __builtin_amdgcn_wmma_f32_16x16x32_bf16(
        /*neg_a=*/false, a, /*neg_b=*/false, b,
        /*c_mod=*/(short)0, c, /*reuse_a=*/false, /*reuse_b=*/false);
}

// A-operand (16x32 bf16, M x K) from row-major [row][stride] storage.
// ISA layout: lane holds row lane%16; K chunks (lane/16)*8 .. +7 and +16.
template <typename PT>
__device__ __forceinline__ v16bf load_a(const PT* rowbase, int stride, int lane) {
    const PT* p = rowbase + (size_t)(lane & 15) * stride + ((lane >> 4) << 3);
    union { v16bf v; v8bf h[2]; } u;
    u.h[0] = *(const v8bf*)(p);
    u.h[1] = *(const v8bf*)(p + 16);
    return u.v;
}

// B-operand (32x16 bf16, K x N) where stored matrix is row-major W[n][k]
// (i.e. column n of B == row n of W).  ISA layout: lane holds column lane%16;
// K chunks (lane/16)*8 .. +7 and +16 (same order as load_a).
template <typename PT>
__device__ __forceinline__ v16bf load_b(const PT* rowbase, int stride, int lane) {
    const PT* p = rowbase + (size_t)(lane & 15) * stride + ((lane >> 4) << 3);
    union { v16bf v; v8bf h[2]; } u;
    u.h[0] = *(const v8bf*)(p);
    u.h[1] = *(const v8bf*)(p + 16);
    return u.v;
}

// CDNA5 async global->LDS copy (16 bytes per active lane), tracked by ASYNCcnt.
__device__ __forceinline__ void async_copy_b128(unsigned lds_off, const void* gptr) {
    asm volatile("global_load_async_to_lds_b128 %0, %1, off"
                 :: "v"(lds_off), "v"(gptr) : "memory");
}
__device__ __forceinline__ void wait_asynccnt0() {
    asm volatile("s_wait_asynccnt 0x0" ::: "memory");
}

// ---------------------------------------------------------------------------
// Prep: fp32 -> bf16 conversions
// ---------------------------------------------------------------------------
__global__ void cvt_bf16_kernel(const float* __restrict__ src,
                                __bf16* __restrict__ dst, int n) {
    for (int i = blockIdx.x * blockDim.x + threadIdx.x; i < n;
         i += gridDim.x * blockDim.x)
        dst[i] = f2bf(src[i]);
}

// sr_w [O=256][I=256][kh=2][kw=2]  ->  [O=256][k=1024], k = (kh*2+kw)*256 + c
__global__ void prep_srw_kernel(const float* __restrict__ sw,
                                __bf16* __restrict__ dst) {
    int i = blockIdx.x * blockDim.x + threadIdx.x;
    if (i < 256 * 1024) {
        int o = i >> 10, k = i & 1023;
        int sl = k >> 8, c = k & 255, kh = sl >> 1, kw = sl & 1;
        dst[i] = f2bf(sw[((o * 256 + c) * 2 + kh) * 2 + kw]);
    }
}

// ---------------------------------------------------------------------------
// q = x @ q_w^T  : [16384 x 256] x [256 x 256] -> q_buf [b][h][n][32] bf16
// block = 256 thr (8 waves); block covers 16 rows x 256 cols; wave: 2 n-tiles
// ---------------------------------------------------------------------------
__global__ __launch_bounds__(256) void gemm_q_kernel(
    const __bf16* __restrict__ xb, const __bf16* __restrict__ wq,
    __bf16* __restrict__ qout) {
    const int tid = threadIdx.x, lane = tid & 31, wave = tid >> 5;
    const int m0 = blockIdx.x * 16;
    v8f acc[2] = {};
    for (int kb = 0; kb < 256; kb += 32) {
        __builtin_prefetch(xb + (size_t)(m0 + 16) * 256 + kb, 0, 1);
        v16bf a = load_a(xb + (size_t)m0 * 256 + kb, 256, lane);
#pragma unroll
        for (int t = 0; t < 2; ++t) {
            int n0 = (wave * 2 + t) * 16;
            v16bf b = load_b(wq + (size_t)n0 * 256 + kb, 256, lane);
            acc[t] = wmma_bf16(a, b, acc[t]);
        }
    }
    const int half = lane >> 4, col = lane & 15;
#pragma unroll
    for (int t = 0; t < 2; ++t) {
        int c = (wave * 2 + t) * 16 + col;
        int h = c >> 5, d = c & 31;
#pragma unroll
        for (int v = 0; v < 8; ++v) {
            int m = m0 + v + 8 * half;
            int b = m >> 12, n = m & 4095;
            qout[((size_t)(b * 8 + h) * 4096 + n) * 32 + d] = f2bf(acc[t][v]);
        }
    }
}

// ---------------------------------------------------------------------------
// SR conv as patch-gather GEMM [4096 x 1024] x [1024 x 256], + bias, + LN,
// -> x_ln [4096 x 256] bf16.  One block per 16 output rows.
// ---------------------------------------------------------------------------
__global__ __launch_bounds__(256) void conv_ln_kernel(
    const __bf16* __restrict__ xb, const __bf16* __restrict__ wsr,
    const float* __restrict__ sr_b, const float* __restrict__ ln_g,
    const float* __restrict__ ln_b, __bf16* __restrict__ xln) {
    __shared__ float sbuf[16][256];
    __shared__ float red[2][16][16];
    __shared__ float rowstat[16][2];
    const int tid = threadIdx.x, lane = tid & 31, wave = tid >> 5;
    const int m0 = blockIdx.x * 16;
    const int r = lane & 15;
    const int m = m0 + r;                 // this lane's A row
    const int b = m >> 10, np = m & 1023;
    const int i = np >> 5, j = np & 31;

    v8f acc[2] = {};
    for (int kb = 0; kb < 1024; kb += 32) {
        int sl = kb >> 8;                 // (kh,kw) slice
        int kh = sl >> 1, kw = sl & 1;
        int c0 = kb & 255;
        int inrow = (2 * i + kh) * 64 + (2 * j + kw);
        const __bf16* p =
            xb + (size_t)(b * 4096 + inrow) * 256 + c0 + ((lane >> 4) << 3);
        union { v16bf v; v8bf h[2]; } ua;
        ua.h[0] = *(const v8bf*)(p);
        ua.h[1] = *(const v8bf*)(p + 16);
#pragma unroll
        for (int t = 0; t < 2; ++t) {
            int n0 = (wave + 8 * t) * 16;
            v16bf bm = load_b(wsr + (size_t)n0 * 1024 + kb, 1024, lane);
            acc[t] = wmma_bf16(ua.v, bm, acc[t]);
        }
    }
    const int half = lane >> 4, col = lane & 15;
#pragma unroll
    for (int t = 0; t < 2; ++t) {
        int c = (wave + 8 * t) * 16 + col;
        float bias = sr_b[c];
#pragma unroll
        for (int v = 0; v < 8; ++v) sbuf[v + 8 * half][c] = acc[t][v] + bias;
    }
    __syncthreads();
    {   // partial sums: 16 threads per row, 16 cols each
        int row = tid >> 4, seg = tid & 15;
        float s = 0.f, s2 = 0.f;
        for (int c = seg * 16; c < seg * 16 + 16; ++c) {
            float x = sbuf[row][c];
            s += x; s2 += x * x;
        }
        red[0][row][seg] = s; red[1][row][seg] = s2;
    }
    __syncthreads();
    if (tid < 16) {
        float s = 0.f, s2 = 0.f;
        for (int g = 0; g < 16; ++g) { s += red[0][tid][g]; s2 += red[1][tid][g]; }
        float mu = s * (1.0f / 256.0f);
        float var = s2 * (1.0f / 256.0f) - mu * mu;
        rowstat[tid][0] = mu;
        rowstat[tid][1] = __frsqrt_rn(var + 1e-5f);
    }
    __syncthreads();
    {
        int row = tid >> 4, seg = tid & 15;
        float mu = rowstat[row][0], rs = rowstat[row][1];
        for (int c = seg * 16; c < seg * 16 + 16; ++c) {
            float y = (sbuf[row][c] - mu) * rs * ln_g[c] + ln_b[c];
            xln[(size_t)(m0 + row) * 256 + c] = f2bf(y);
        }
    }
}

// ---------------------------------------------------------------------------
// kv = x_ln @ kv_w^T : [4096 x 256] x [256 x 512]
// cols 0..255  -> K  stored [b][h][n'][32]     (row-major per head)
// cols 256..511-> V  stored [b][h][32][1024]   (transposed per head)
// ---------------------------------------------------------------------------
__global__ __launch_bounds__(256) void gemm_kv_kernel(
    const __bf16* __restrict__ xln, const __bf16* __restrict__ wkv,
    __bf16* __restrict__ kbuf, __bf16* __restrict__ vtbuf) {
    const int tid = threadIdx.x, lane = tid & 31, wave = tid >> 5;
    const int m0 = blockIdx.x * 16;
    v8f acc[4] = {};
    for (int kb = 0; kb < 256; kb += 32) {
        v16bf a = load_a(xln + (size_t)m0 * 256 + kb, 256, lane);
#pragma unroll
        for (int t = 0; t < 4; ++t) {
            int n0 = (wave * 4 + t) * 16;
            v16bf b = load_b(wkv + (size_t)n0 * 256 + kb, 256, lane);
            acc[t] = wmma_bf16(a, b, acc[t]);
        }
    }
    const int half = lane >> 4, col = lane & 15;
#pragma unroll
    for (int t = 0; t < 4; ++t) {
        int c = (wave * 4 + t) * 16 + col;
#pragma unroll
        for (int v = 0; v < 8; ++v) {
            int m = m0 + v + 8 * half;
            int bq = m >> 10, np = m & 1023;
            __bf16 val = f2bf(acc[t][v]);
            if (c < 256) {
                int h = c >> 5, d = c & 31;
                kbuf[((size_t)(bq * 8 + h) * 1024 + np) * 32 + d] = val;
            } else {
                int cc = c - 256, h = cc >> 5, d = cc & 31;
                vtbuf[((size_t)(bq * 8 + h) * 32 + d) * 1024 + np] = val;
            }
        }
    }
}

// ---------------------------------------------------------------------------
// Flash attention per (b,h).  grid = (32 q-blocks of 128 rows, 32 bh).
// K (1024x32) and V^T (32x1024) staged whole in LDS (128 KB of the 320 KB WGP
// LDS) with CDNA5 async global->LDS copies (ASYNCcnt).  Each wave owns 16 Q
// rows; online softmax with shfl_xor reductions over 16-lane halves; P is
// converted C-layout -> A-layout via a per-wave LDS staging tile.
// ---------------------------------------------------------------------------
__global__ __launch_bounds__(256) void attn_kernel(
    const __bf16* __restrict__ qbuf, const __bf16* __restrict__ kbuf,
    const __bf16* __restrict__ vtbuf, __bf16* __restrict__ obuf) {
    extern __shared__ __bf16 smem[];
    __bf16* Ks = smem;                    // [1024][32]
    __bf16* Vs = smem + 1024 * 32;        // [32][1024]
    __bf16* Ps = smem + 2 * 1024 * 32;    // 8 waves x [16][32]

    const int tid = threadIdx.x, lane = tid & 31, wave = tid >> 5;
    const int bh = blockIdx.y;
    const int q0 = blockIdx.x * 128 + wave * 16;

    // async stage of K and V^T (64 KB each) directly into LDS
    const __bf16* kg = kbuf + (size_t)bh * 1024 * 32;
    const __bf16* vg = vtbuf + (size_t)bh * 1024 * 32;
    for (int it = 0; it < 16; ++it) {
        int idx = (it * 256 + tid) * 8;   // 8 bf16 = 16 B per lane
        async_copy_b128((unsigned)(unsigned long long)(Ks + idx), kg + idx);
        async_copy_b128((unsigned)(unsigned long long)(Vs + idx), vg + idx);
    }

    // overlap: load this wave's Q tile while the async fill is in flight
    v16bf qa = load_a(qbuf + ((size_t)bh * 4096 + q0) * 32, 32, lane);

    wait_asynccnt0();
    __syncthreads();

    v8f O0 = {}, O1 = {};
    float mrow[8], lrow[8];
#pragma unroll
    for (int v = 0; v < 8; ++v) { mrow[v] = -1e30f; lrow[v] = 0.f; }
    const float scale = 0.17677669529663687f;   // 32^-0.5
    __bf16* Pw = Ps + wave * 16 * 32;
    const int half = lane >> 4, col = lane & 15;

    for (int pr = 0; pr < 32; ++pr) {
        const int s0 = pr * 32;                 // 32 seq positions per step
        v16bf kb0 = load_b(Ks + (size_t)s0 * 32, 32, lane);
        v16bf kb1 = load_b(Ks + (size_t)(s0 + 16) * 32, 32, lane);
        v8f zf = {};
        v8f S0 = wmma_bf16(qa, kb0, zf);
        v8f S1 = wmma_bf16(qa, kb1, zf);
#pragma unroll
        for (int v = 0; v < 8; ++v) {
            float a0 = S0[v] * scale, a1 = S1[v] * scale;
            float mx = fmaxf(a0, a1);
            mx = fmaxf(mx, __shfl_xor(mx, 1, 32));
            mx = fmaxf(mx, __shfl_xor(mx, 2, 32));
            mx = fmaxf(mx, __shfl_xor(mx, 4, 32));
            mx = fmaxf(mx, __shfl_xor(mx, 8, 32));
            float mnew = fmaxf(mrow[v], mx);
            float alpha = __expf(mrow[v] - mnew);
            mrow[v] = mnew;
            float p0 = __expf(a0 - mnew), p1 = __expf(a1 - mnew);
            float rs = p0 + p1;
            rs += __shfl_xor(rs, 1, 32);
            rs += __shfl_xor(rs, 2, 32);
            rs += __shfl_xor(rs, 4, 32);
            rs += __shfl_xor(rs, 8, 32);
            lrow[v] = lrow[v] * alpha + rs;
            O0[v] *= alpha;
            O1[v] *= alpha;
            int prow = v + 8 * half;
            Pw[prow * 32 + col]      = f2bf(p0);
            Pw[prow * 32 + 16 + col] = f2bf(p1);
        }
        // P (16x32) back as A-operand; V^T rows as B-operand
        v16bf pa  = load_a(Pw, 32, lane);
        v16bf vb0 = load_b(Vs + s0, 1024, lane);                 // d 0..15
        v16bf vb1 = load_b(Vs + 16 * 1024 + s0, 1024, lane);     // d 16..31
        O0 = wmma_bf16(pa, vb0, O0);
        O1 = wmma_bf16(pa, vb1, O1);
    }

    const int b = bh >> 3, h = bh & 7;
#pragma unroll
    for (int v = 0; v < 8; ++v) {
        float inv = 1.0f / lrow[v];
        int row = q0 + v + 8 * half;
        size_t base = ((size_t)(b * 4096 + row)) * 256 + h * 32;
        obuf[base + col]      = f2bf(O0[v] * inv);
        obuf[base + 16 + col] = f2bf(O1[v] * inv);
    }
}

// ---------------------------------------------------------------------------
// out = attn_out @ proj_w^T + proj_b : [16384 x 256] x [256 x 256] -> fp32
// ---------------------------------------------------------------------------
__global__ __launch_bounds__(256) void gemm_proj_kernel(
    const __bf16* __restrict__ ob, const __bf16* __restrict__ wp,
    const float* __restrict__ pb, float* __restrict__ out) {
    const int tid = threadIdx.x, lane = tid & 31, wave = tid >> 5;
    const int m0 = blockIdx.x * 16;
    v8f acc[2] = {};
    for (int kb = 0; kb < 256; kb += 32) {
        v16bf a = load_a(ob + (size_t)m0 * 256 + kb, 256, lane);
#pragma unroll
        for (int t = 0; t < 2; ++t) {
            int n0 = (wave * 2 + t) * 16;
            v16bf b = load_b(wp + (size_t)n0 * 256 + kb, 256, lane);
            acc[t] = wmma_bf16(a, b, acc[t]);
        }
    }
    const int half = lane >> 4, col = lane & 15;
#pragma unroll
    for (int t = 0; t < 2; ++t) {
        int c = (wave * 2 + t) * 16 + col;
        float bias = pb[c];
#pragma unroll
        for (int v = 0; v < 8; ++v) {
            int m = m0 + v + 8 * half;
            out[(size_t)m * 256 + c] = acc[t][v] + bias;
        }
    }
}

// ---------------------------------------------------------------------------
extern "C" void kernel_launch(void* const* d_in, const int* in_sizes, int n_in,
                              void* d_out, int out_size, void* d_ws, size_t ws_size,
                              hipStream_t stream) {
    (void)in_sizes; (void)n_in; (void)out_size; (void)ws_size;
    const float* x      = (const float*)d_in[0];
    // d_in[1]=H(64), d_in[2]=W(64) -- hardcoded
    const float* sr_w   = (const float*)d_in[3];
    const float* sr_b   = (const float*)d_in[4];
    const float* ln_g   = (const float*)d_in[5];
    const float* ln_b   = (const float*)d_in[6];
    const float* q_w    = (const float*)d_in[7];
    const float* kv_w   = (const float*)d_in[8];
    const float* proj_w = (const float*)d_in[9];
    const float* proj_b = (const float*)d_in[10];
    float* out = (float*)d_out;

    char* ws = (char*)d_ws;
    size_t off = 0;
    auto alloc = [&](size_t bytes) {
        char* p = ws + off;
        off += (bytes + 255) & ~(size_t)255;
        return p;
    };
    __bf16* xb    = (__bf16*)alloc(16384ull * 256 * 2);   // x bf16
    __bf16* qwb   = (__bf16*)alloc(256ull * 256 * 2);
    __bf16* kvwb  = (__bf16*)alloc(512ull * 256 * 2);
    __bf16* pwb   = (__bf16*)alloc(256ull * 256 * 2);
    __bf16* srwb  = (__bf16*)alloc(256ull * 1024 * 2);    // rearranged conv W
    __bf16* qbuf  = (__bf16*)alloc(16384ull * 256 * 2);   // [b][h][n][32]
    __bf16* xln   = (__bf16*)alloc(4096ull * 256 * 2);
    __bf16* kbuf  = (__bf16*)alloc(4096ull * 256 * 2);    // [b][h][n'][32]
    __bf16* vtbuf = (__bf16*)alloc(4096ull * 256 * 2);    // [b][h][32][n']
    __bf16* obuf  = (__bf16*)alloc(16384ull * 256 * 2);   // attn out [b][n][c]

    cvt_bf16_kernel<<<4096, 256, 0, stream>>>(x, xb, 16384 * 256);
    cvt_bf16_kernel<<<256, 256, 0, stream>>>(q_w, qwb, 256 * 256);
    cvt_bf16_kernel<<<512, 256, 0, stream>>>(kv_w, kvwb, 512 * 256);
    cvt_bf16_kernel<<<256, 256, 0, stream>>>(proj_w, pwb, 256 * 256);
    prep_srw_kernel<<<1024, 256, 0, stream>>>(sr_w, srwb);

    gemm_q_kernel<<<1024, 256, 0, stream>>>(xb, qwb, qbuf);
    conv_ln_kernel<<<256, 256, 0, stream>>>(xb, srwb, sr_b, ln_g, ln_b, xln);
    gemm_kv_kernel<<<256, 256, 0, stream>>>(xln, kvwb, kbuf, vtbuf);

    dim3 agrid(32, 32);   // (q-blocks of 128 rows, b*h)
    size_t smem_attn = (2ull * 1024 * 32 + 8ull * 16 * 32) * sizeof(__bf16); // 136 KB
    attn_kernel<<<agrid, 256, smem_attn, stream>>>(qbuf, kbuf, vtbuf, obuf);

    gemm_proj_kernel<<<1024, 256, 0, stream>>>(obuf, pwb, proj_b, out);
}
